// SlotAttentionEncoder_24120536334725
// MI455X (gfx1250) — matrix operand order repaired and checked by reading, not compiled, not run on hardware
//
#include <hip/hip_runtime.h>
#include <hip/hip_bf16.h>

// ---------------------------------------------------------------------------
// Types / WMMA helpers (CDNA5 gfx1250, wave32)
// ---------------------------------------------------------------------------
typedef __attribute__((ext_vector_type(16))) __bf16 bf16x16;
typedef __attribute__((ext_vector_type(8)))  __bf16 bf16x8;
typedef __attribute__((ext_vector_type(8)))  float  f32x8;

#define WMMA_BF16(a, b, c) \
  __builtin_amdgcn_wmma_f32_16x16x32_bf16(false, (a), false, (b), (short)0, (c), false, false)

// CDNA5 async global->LDS copy (ASYNCcnt path), 16B per lane.
__device__ __forceinline__ void async_b128(void* lds, const void* g) {
  asm volatile("global_load_async_to_lds_b128 %0, %1, off"
               :: "v"((unsigned)(uintptr_t)lds),
                  "v"((unsigned long long)(uintptr_t)g) : "memory");
}
__device__ __forceinline__ void wait_async0() {
  asm volatile("s_wait_asynccnt 0" ::: "memory");
}

// A-matrix 16x32 bf16 fragment from row-major A[.., lda].
// Lane L: row = row_base + (L&15); e<8 -> K=k0+off+e, e>=8 -> K=k0+off+16+(e-8),
// off = (L>=16)?8:0.  (ISA 7.12.2)
__device__ __forceinline__ bf16x16 frag_a(const __bf16* A, int lda, int row_base,
                                          int k_base, int lane) {
  int r    = row_base + (lane & 15);
  int koff = k_base + ((lane >> 4) << 3);
  const __bf16* p = A + (size_t)r * lda + koff;
  bf16x8 lo = *(const bf16x8*)(p);
  bf16x8 hi = *(const bf16x8*)(p + 16);
  bf16x16 f;
#pragma unroll
  for (int i = 0; i < 8; ++i) { f[i] = lo[i]; f[i + 8] = hi[i]; }
  return f;
}

// B-matrix 32x16 bf16 fragment where B[k][n] = W[n][k] (W row-major [N,ldw]).
// Lane L: col n = n_base + (L&15); e<8 -> K=k0+off+e, e>=8 -> K=k0+off+16+(e-8), off = (L>=16)?8:0.
__device__ __forceinline__ bf16x16 frag_b(const __bf16* W, int ldw, int n_base,
                                          int k_base, int lane) {
  int n    = n_base + (lane & 15);
  int koff = k_base + ((lane >> 4) << 3);
  const __bf16* p = W + (size_t)n * ldw + koff;
  bf16x8 lo = *(const bf16x8*)(p);
  bf16x8 hi = *(const bf16x8*)(p + 16);
  bf16x16 f;
#pragma unroll
  for (int i = 0; i < 8; ++i) { f[i] = lo[i]; f[i + 8] = hi[i]; }
  return f;
}

// A fragment built on the fly from f32 data (LDS attention matrix).
__device__ __forceinline__ bf16x16 frag_a_f32(const float* A, int lda,
                                              int k_base, int lane) {
  int r    = lane & 15;
  int koff = k_base + ((lane >> 4) << 3);
  const float* p = A + (size_t)r * lda + koff;
  bf16x16 f;
#pragma unroll
  for (int i = 0; i < 8; ++i) { f[i] = (__bf16)p[i]; f[i + 8] = (__bf16)p[i + 16]; }
  return f;
}

__device__ __forceinline__ float sigmoidf_(float x) { return 1.0f / (1.0f + __expf(-x)); }

#define WST 520   // bf16 stride for LDS-staged weight tiles (16B aligned, bank-spread)

// ---------------------------------------------------------------------------
// f32 -> bf16 convert (weights)
// ---------------------------------------------------------------------------
__global__ void cvt_bf16_kernel(const float* __restrict__ in, __bf16* __restrict__ out, int n) {
  int i = blockIdx.x * 256 + threadIdx.x;
  if (i < n) out[i] = (__bf16)in[i];
}

// ---------------------------------------------------------------------------
// LayerNorm over rows of length 512, one wave per row, writes bf16.
// ---------------------------------------------------------------------------
__launch_bounds__(256)
__global__ void ln_kernel(const float* __restrict__ X, const float* __restrict__ g,
                          const float* __restrict__ bta, __bf16* __restrict__ outB, int rows) {
  int row  = blockIdx.x * 8 + (threadIdx.x >> 5);
  int lane = threadIdx.x & 31;
  if (row >= rows) return;
  const float* x = X + (size_t)row * 512 + lane * 16;
  float v[16];
#pragma unroll
  for (int c = 0; c < 4; ++c) {
    float4 t = *(const float4*)(x + c * 4);
    v[c * 4 + 0] = t.x; v[c * 4 + 1] = t.y; v[c * 4 + 2] = t.z; v[c * 4 + 3] = t.w;
  }
  float s = 0.f;
#pragma unroll
  for (int i = 0; i < 16; ++i) s += v[i];
#pragma unroll
  for (int o = 16; o > 0; o >>= 1) s += __shfl_xor(s, o, 32);
  float m = s * (1.0f / 512.0f);
  float vs = 0.f;
#pragma unroll
  for (int i = 0; i < 16; ++i) { float d = v[i] - m; vs += d * d; }
#pragma unroll
  for (int o = 16; o > 0; o >>= 1) vs += __shfl_xor(vs, o, 32);
  float rstd = rsqrtf(vs * (1.0f / 512.0f) + 1e-8f);

  bf16x8 o0, o1;
#pragma unroll
  for (int i = 0; i < 8; ++i) {
    int c0 = lane * 16 + i, c1 = lane * 16 + 8 + i;
    o0[i] = (__bf16)((v[i]     - m) * rstd * g[c0] + bta[c0]);
    o1[i] = (__bf16)((v[i + 8] - m) * rstd * g[c1] + bta[c1]);
  }
  __bf16* op = outB + (size_t)row * 512 + lane * 16;
  *(bf16x8*)(op)     = o0;
  *(bf16x8*)(op + 8) = o1;
}

// ---------------------------------------------------------------------------
// Dual GEMM: K = A @ Wk^T, V = A @ Wv^T.  A:[M,512] bf16. Wave tile 16x64,
// block 8 waves -> 128x64. grid = (512/64, M/128).
// Weight tiles async-staged to LDS once per block; B fragments then come
// from LDS at ds latency. A prefetched one k-step ahead from global.
// ---------------------------------------------------------------------------
#define KV_SMEM_BYTES (2 * 64 * WST * 2)

__launch_bounds__(256)
__global__ void kv_gemm_kernel(const __bf16* __restrict__ A,
                               const __bf16* __restrict__ Wk, const __bf16* __restrict__ Wv,
                               __bf16* __restrict__ Kout, __bf16* __restrict__ Vout) {
  extern __shared__ char smem[];
  __bf16* WkS = (__bf16*)smem;                       // [64][WST]
  __bf16* WvS = (__bf16*)(smem + 64 * WST * 2);      // [64][WST]

  int tid = threadIdx.x, lane = tid & 31, w = tid >> 5;
  int mbase = blockIdx.y * 128 + w * 16;
  int nbase = blockIdx.x * 64;

  // stage weight tiles (rows nbase..nbase+63, full K) via async copy
  for (int c = tid; c < 4096; c += 256) {            // 8-bf16 chunks
    int r = c >> 6, cb = (c & 63) << 3;
    async_b128(WkS + r * WST + cb, Wk + (size_t)(nbase + r) * 512 + cb);
    async_b128(WvS + r * WST + cb, Wv + (size_t)(nbase + r) * 512 + cb);
  }
  wait_async0();
  __syncthreads();

  f32x8 ak[4] = {}, av[4] = {};
  bf16x16 a = frag_a(A, 512, mbase, 0, lane);
#pragma unroll
  for (int k0 = 0; k0 < 512; k0 += 32) {
    bf16x16 an = (k0 < 480) ? frag_a(A, 512, mbase, k0 + 32, lane) : a;
#pragma unroll
    for (int t = 0; t < 4; ++t) ak[t] = WMMA_BF16(a, frag_b(WkS, WST, t * 16, k0, lane), ak[t]);
#pragma unroll
    for (int t = 0; t < 4; ++t) av[t] = WMMA_BF16(a, frag_b(WvS, WST, t * 16, k0, lane), av[t]);
    a = an;
  }
  int rof = (lane >> 4) << 3, c0 = lane & 15;
#pragma unroll
  for (int t = 0; t < 4; ++t) {
    int col = nbase + t * 16 + c0;
#pragma unroll
    for (int j = 0; j < 8; ++j) {
      size_t idx = (size_t)(mbase + rof + j) * 512 + col;
      Kout[idx] = (__bf16)ak[t][j];
      Vout[idx] = (__bf16)av[t][j];
    }
  }
}

// ---------------------------------------------------------------------------
// Generic WMMA GEMM: out = act(A @ W^T + bias + residual), A:[M,512] bf16,
// W:[N,512] bf16. Optional f32 and bf16 outputs. grid = (N/64, M/128).
// Weight tile async-staged to LDS; A prefetched one k-step ahead.
// ---------------------------------------------------------------------------
#define GEMM_SMEM_BYTES (64 * WST * 2)

__launch_bounds__(256)
__global__ void gemm_kernel(const __bf16* __restrict__ A, const __bf16* __restrict__ W,
                            const float* __restrict__ bias, const float* __restrict__ residual,
                            float* __restrict__ outF, __bf16* __restrict__ outB,
                            int N, int relu) {
  extern __shared__ char smem[];
  __bf16* WS = (__bf16*)smem;                        // [64][WST]

  int tid = threadIdx.x, lane = tid & 31, w = tid >> 5;
  int mbase = blockIdx.y * 128 + w * 16;
  int nbase = blockIdx.x * 64;

  for (int c = tid; c < 4096; c += 256) {
    int r = c >> 6, cb = (c & 63) << 3;
    async_b128(WS + r * WST + cb, W + (size_t)(nbase + r) * 512 + cb);
  }
  wait_async0();
  __syncthreads();

  f32x8 acc[4] = {};
  bf16x16 a = frag_a(A, 512, mbase, 0, lane);
#pragma unroll
  for (int k0 = 0; k0 < 512; k0 += 32) {
    bf16x16 an = (k0 < 480) ? frag_a(A, 512, mbase, k0 + 32, lane) : a;
#pragma unroll
    for (int t = 0; t < 4; ++t) acc[t] = WMMA_BF16(a, frag_b(WS, WST, t * 16, k0, lane), acc[t]);
    a = an;
  }

  int rof = (lane >> 4) << 3, c0 = lane & 15;
#pragma unroll
  for (int t = 0; t < 4; ++t) {
    int col = nbase + t * 16 + c0;
    float bv = bias ? bias[col] : 0.0f;
#pragma unroll
    for (int j = 0; j < 8; ++j) {
      int row = mbase + rof + j;
      float x = acc[t][j] + bv;
      if (residual) x += residual[(size_t)row * N + col];
      if (relu) x = fmaxf(x, 0.0f);
      if (outF) outF[(size_t)row * N + col] = x;
      if (outB) outB[(size_t)row * N + col] = (__bf16)x;
    }
  }
}

// ---------------------------------------------------------------------------
// Fused attention, one block per batch.
//   q staged to LDS via async global->LDS copy
//   logits[16,1024] = (q @ k^T) * D^-0.5     (WMMA, logits in LDS, k prefetched)
//   row softmax over N, column-normalize over S (in place, f32)
//   updates[16,512] = attn @ v               (WMMA, ping-pong transposed v in LDS)
// ---------------------------------------------------------------------------
#define LST   1028   // f32 stride, logits rows (16B aligned, bank-spread)
#define QST   520    // bf16 stride, staged q  (16B aligned, bank-spread)
#define VTST  40     // bf16 stride, vT rows   (80B -> 16B aligned)
#define LOGITS_BYTES (16 * LST * 4)
#define QS_BYTES     (16 * QST * 2)
#define VT_BYTES     (512 * VTST * 2)
#define ATTN_SMEM_BYTES (LOGITS_BYTES + QS_BYTES + 2 * VT_BYTES)

__launch_bounds__(256)
__global__ void attn_kernel(const __bf16* __restrict__ qB, const __bf16* __restrict__ kB,
                            const __bf16* __restrict__ vB, __bf16* __restrict__ updB) {
  extern __shared__ char smem[];
  float*  logits = (float*)smem;                                  // [16][LST]
  __bf16* qS     = (__bf16*)(smem + LOGITS_BYTES);                // [16][QST]
  __bf16* vT0    = (__bf16*)(smem + LOGITS_BYTES + QS_BYTES);     // [512][VTST]
  __bf16* vT1    = (__bf16*)(smem + LOGITS_BYTES + QS_BYTES + VT_BYTES);

  int b = blockIdx.x;
  int tid = threadIdx.x, lane = tid & 31, w = tid >> 5;
  const __bf16* q  = qB + (size_t)b * 16 * 512;
  const __bf16* kb = kB + (size_t)b * 1024 * 512;
  const __bf16* vb = vB + (size_t)b * 1024 * 512;

  // ---- Phase 0: async-copy q (16x512 bf16) into LDS, padded stride.
  for (int c = tid; c < 1024; c += 256) {
    int r = c >> 6, cb = (c & 63) << 3;
    async_b128(qS + r * QST + cb, q + (size_t)r * 512 + cb);
  }
  wait_async0();
  __syncthreads();

  // ---- Phase 1: logits tiles; wave w owns columns [w*128, w*128+128).
  {
    f32x8 acc[8] = {};
    int nbase = w * 128;
    bf16x16 bcur[8];
#pragma unroll
    for (int t = 0; t < 8; ++t) bcur[t] = frag_b(kb, 512, nbase + t * 16, 0, lane);
#pragma unroll
    for (int k0 = 0; k0 < 512; k0 += 32) {
      bf16x16 a = frag_a(qS, QST, 0, k0, lane);  // LDS, low latency
      bf16x16 bn[8];
      if (k0 < 480) {
#pragma unroll
        for (int t = 0; t < 8; ++t) bn[t] = frag_b(kb, 512, nbase + t * 16, k0 + 32, lane);
      }
#pragma unroll
      for (int t = 0; t < 8; ++t) acc[t] = WMMA_BF16(a, bcur[t], acc[t]);
      if (k0 < 480) {
#pragma unroll
        for (int t = 0; t < 8; ++t) bcur[t] = bn[t];
      }
    }
    const float scale = 0.044194173824159216f;  // 512^-0.5
    int rof = (lane >> 4) << 3, c0 = lane & 15;
#pragma unroll
    for (int t = 0; t < 8; ++t)
#pragma unroll
      for (int j = 0; j < 8; ++j)
        logits[(rof + j) * LST + nbase + t * 16 + c0] = acc[t][j] * scale;
  }
  __syncthreads();

  // ---- Phase 2: row softmax over N=1024 (wave w handles rows 2w, 2w+1).
#pragma unroll
  for (int rr = 0; rr < 2; ++rr) {
    float* row = logits + (w * 2 + rr) * LST;
    float mx = -3.402823466e38f;
    for (int n = lane; n < 1024; n += 32) mx = fmaxf(mx, row[n]);
#pragma unroll
    for (int o = 16; o > 0; o >>= 1) mx = fmaxf(mx, __shfl_xor(mx, o, 32));
    float sum = 0.f;
    for (int n = lane; n < 1024; n += 32) { float e = __expf(row[n] - mx); row[n] = e; sum += e; }
#pragma unroll
    for (int o = 16; o > 0; o >>= 1) sum += __shfl_xor(sum, o, 32);
    float inv = 1.0f / sum;
    for (int n = lane; n < 1024; n += 32) row[n] *= inv;
  }
  __syncthreads();

  // ---- Phase 3: normalize each column over the 16 slots (in place, f32).
  for (int n = tid; n < 1024; n += 256) {
    float cs = 0.f;
#pragma unroll
    for (int s = 0; s < 16; ++s) cs += logits[s * LST + n];
    float inv = 1.0f / (cs + 1e-8f);
#pragma unroll
    for (int s = 0; s < 16; ++s) logits[s * LST + n] *= inv;
  }
  __syncthreads();

  // ---- Phase 4: updates = attn @ v. Ping-pong LDS v^T blocks; the next
  // block's global loads are issued into registers before the WMMAs, and
  // committed to the alternate buffer after, one barrier per step.
  for (int c = tid; c < 2048; c += 256) {          // stage block 0 -> vT0
    int r = c >> 6, cb = (c & 63) << 3;
    bf16x8 vv = *(const bf16x8*)(vb + (size_t)r * 512 + cb);
#pragma unroll
    for (int qq = 0; qq < 8; ++qq) vT0[(cb + qq) * VTST + r] = vv[qq];
  }
  __syncthreads();

  f32x8 uacc[4] = {};
  int dbase = w * 64;
  for (int kb2 = 0; kb2 < 1024; kb2 += 32) {
    __bf16* cur = (kb2 & 32) ? vT1 : vT0;
    __bf16* nxt = (kb2 & 32) ? vT0 : vT1;
    bool more = (kb2 + 32) < 1024;
    bf16x8 vv[8];
    if (more) {
#pragma unroll
      for (int j = 0; j < 8; ++j) {
        int c = tid + j * 256;
        int r = c >> 6, cb = (c & 63) << 3;
        vv[j] = *(const bf16x8*)(vb + (size_t)(kb2 + 32 + r) * 512 + cb);
      }
    }
    bf16x16 a = frag_a_f32(logits, LST, kb2, lane);
#pragma unroll
    for (int t = 0; t < 4; ++t) {
      bf16x16 bf = frag_b(cur, VTST, dbase + t * 16, 0, lane);
      uacc[t] = WMMA_BF16(a, bf, uacc[t]);
    }
    if (more) {
#pragma unroll
      for (int j = 0; j < 8; ++j) {
        int c = tid + j * 256;
        int r = c >> 6, cb = (c & 63) << 3;
#pragma unroll
        for (int qq = 0; qq < 8; ++qq) nxt[(cb + qq) * VTST + r] = vv[j][qq];
      }
    }
    __syncthreads();
  }
  int rof = (lane >> 4) << 3, c0 = lane & 15;
#pragma unroll
  for (int t = 0; t < 4; ++t)
#pragma unroll
    for (int j = 0; j < 8; ++j)
      updB[(size_t)(b * 16 + rof + j) * 512 + dbase + t * 16 + c0] = (__bf16)uacc[t][j];
}

// ---------------------------------------------------------------------------
// slots = mu + exp(log_sigma) * noise   (writes f32 + bf16)
// ---------------------------------------------------------------------------
__global__ void slots_init_kernel(const float* __restrict__ noise, const float* __restrict__ mu,
                                  const float* __restrict__ lsig, float* __restrict__ sF,
                                  __bf16* __restrict__ sB, int n) {
  int i = blockIdx.x * 256 + threadIdx.x;
  if (i >= n) return;
  int d = i & 511;
  float v = mu[d] + __expf(lsig[d]) * noise[i];
  sF[i] = v; sB[i] = (__bf16)v;
}

// ---------------------------------------------------------------------------
// GRU gates (r,z,n ordering, PyTorch GRUCell). In-place h update.
// ---------------------------------------------------------------------------
__global__ void gru_gates_kernel(const float* __restrict__ gi, const float* __restrict__ gh,
                                 float* __restrict__ slotsF, __bf16* __restrict__ slotsB, int n) {
  int i = blockIdx.x * 256 + threadIdx.x;
  if (i >= n) return;
  int row = i >> 9, col = i & 511;
  size_t base = (size_t)row * 1536 + col;
  float ir = gi[base], iz = gi[base + 512], inn = gi[base + 1024];
  float hr = gh[base], hz = gh[base + 512], hn  = gh[base + 1024];
  float r  = sigmoidf_(ir + hr);
  float z  = sigmoidf_(iz + hz);
  float nn = tanhf(inn + r * hn);
  float h  = slotsF[i];
  float o  = (1.0f - z) * nn + z * h;
  slotsF[i] = o; slotsB[i] = (__bf16)o;
}

// ---------------------------------------------------------------------------
// Host-side launcher
// ---------------------------------------------------------------------------
extern "C" void kernel_launch(void* const* d_in, const int* in_sizes, int n_in,
                              void* d_out, int out_size, void* d_ws, size_t ws_size,
                              hipStream_t stream) {
  (void)in_sizes; (void)n_in; (void)out_size; (void)ws_size;
  const float* inputs     = (const float*)d_in[0];
  const float* noise      = (const float*)d_in[1];
  const float* slot_mu    = (const float*)d_in[2];
  const float* slot_lsig  = (const float*)d_in[3];
  const float* ln_in_g    = (const float*)d_in[4];
  const float* ln_in_b    = (const float*)d_in[5];
  const float* ln_sl_g    = (const float*)d_in[6];
  const float* ln_sl_b    = (const float*)d_in[7];
  const float* ln_mlp_g   = (const float*)d_in[8];
  const float* ln_mlp_b   = (const float*)d_in[9];
  const float* Wq         = (const float*)d_in[10];
  const float* Wk         = (const float*)d_in[11];
  const float* Wv         = (const float*)d_in[12];
  const float* gru_w_ih   = (const float*)d_in[13];
  const float* gru_w_hh   = (const float*)d_in[14];
  const float* gru_b_ih   = (const float*)d_in[15];
  const float* gru_b_hh   = (const float*)d_in[16];
  const float* mlp_w1     = (const float*)d_in[17];
  const float* mlp_b1     = (const float*)d_in[18];
  const float* mlp_w2     = (const float*)d_in[19];
  const float* mlp_b2     = (const float*)d_in[20];
  float* out = (float*)d_out;

  const int B = 64, N = 1024, D = 512, S = 16;
  const int BN = B * N;            // 65536
  const int BS = B * S;            // 1024

  // workspace carve
  char* base = (char*)d_ws; size_t off = 0;
  auto alloc = [&](size_t bytes) -> void* {
    void* r = base + off; off = (off + bytes + 255) & ~(size_t)255; return r;
  };
  __bf16* WqB  = (__bf16*)alloc((size_t)D * D * 2);
  __bf16* WkB  = (__bf16*)alloc((size_t)D * D * 2);
  __bf16* WvB  = (__bf16*)alloc((size_t)D * D * 2);
  __bf16* WihB = (__bf16*)alloc((size_t)3 * D * D * 2);
  __bf16* WhhB = (__bf16*)alloc((size_t)3 * D * D * 2);
  __bf16* W1B  = (__bf16*)alloc((size_t)D * D * 2);
  __bf16* W2B  = (__bf16*)alloc((size_t)D * D * 2);
  __bf16* xnB  = (__bf16*)alloc((size_t)BN * D * 2);
  __bf16* kBuf = (__bf16*)alloc((size_t)BN * D * 2);
  __bf16* vBuf = (__bf16*)alloc((size_t)BN * D * 2);
  float*  slotsF = (float*)alloc((size_t)BS * D * 4);
  __bf16* slotsB = (__bf16*)alloc((size_t)BS * D * 2);
  __bf16* snB    = (__bf16*)alloc((size_t)BS * D * 2);
  __bf16* qB     = (__bf16*)alloc((size_t)BS * D * 2);
  __bf16* updB   = (__bf16*)alloc((size_t)BS * D * 2);
  float*  giF    = (float*)alloc((size_t)BS * 3 * D * 4);
  float*  ghF    = (float*)alloc((size_t)BS * 3 * D * 4);
  __bf16* sn2B   = (__bf16*)alloc((size_t)BS * D * 2);
  __bf16* hB     = (__bf16*)alloc((size_t)BS * D * 2);

  // weights -> bf16
  cvt_bf16_kernel<<<(D * D + 255) / 256, 256, 0, stream>>>(Wq, WqB, D * D);
  cvt_bf16_kernel<<<(D * D + 255) / 256, 256, 0, stream>>>(Wk, WkB, D * D);
  cvt_bf16_kernel<<<(D * D + 255) / 256, 256, 0, stream>>>(Wv, WvB, D * D);
  cvt_bf16_kernel<<<(3 * D * D + 255) / 256, 256, 0, stream>>>(gru_w_ih, WihB, 3 * D * D);
  cvt_bf16_kernel<<<(3 * D * D + 255) / 256, 256, 0, stream>>>(gru_w_hh, WhhB, 3 * D * D);
  cvt_bf16_kernel<<<(D * D + 255) / 256, 256, 0, stream>>>(mlp_w1, W1B, D * D);
  cvt_bf16_kernel<<<(D * D + 255) / 256, 256, 0, stream>>>(mlp_w2, W2B, D * D);

  // LN(inputs) -> bf16; then k,v projections (dual GEMM, one pass over xn)
  ln_kernel<<<BN / 8, 256, 0, stream>>>(inputs, ln_in_g, ln_in_b, xnB, BN);
  kv_gemm_kernel<<<dim3(D / 64, BN / 128), 256, KV_SMEM_BYTES, stream>>>(xnB, WkB, WvB,
                                                                         kBuf, vBuf);

  // slot init
  slots_init_kernel<<<(BS * D + 255) / 256, 256, 0, stream>>>(noise, slot_mu, slot_lsig,
                                                              slotsF, slotsB, BS * D);

  for (int it = 0; it < 3; ++it) {
    // q = LN(slots) @ Wq^T
    ln_kernel<<<BS / 8, 256, 0, stream>>>(slotsF, ln_sl_g, ln_sl_b, snB, BS);
    gemm_kernel<<<dim3(D / 64, BS / 128), 256, GEMM_SMEM_BYTES, stream>>>(
        snB, WqB, nullptr, nullptr, nullptr, qB, D, 0);

    // fused attention + updates
    attn_kernel<<<B, 256, ATTN_SMEM_BYTES, stream>>>(qB, kBuf, vBuf, updB);

    // GRU: gi = updates @ w_ih^T + b_ih ; gh = slots_prev @ w_hh^T + b_hh
    gemm_kernel<<<dim3(3 * D / 64, BS / 128), 256, GEMM_SMEM_BYTES, stream>>>(
        updB, WihB, gru_b_ih, nullptr, giF, nullptr, 3 * D, 0);
    gemm_kernel<<<dim3(3 * D / 64, BS / 128), 256, GEMM_SMEM_BYTES, stream>>>(
        slotsB, WhhB, gru_b_hh, nullptr, ghF, nullptr, 3 * D, 0);
    gru_gates_kernel<<<(BS * D + 255) / 256, 256, 0, stream>>>(giF, ghF, slotsF, slotsB, BS * D);

    // MLP with residual: slots += relu(LN(slots) @ w1^T + b1) @ w2^T + b2
    ln_kernel<<<BS / 8, 256, 0, stream>>>(slotsF, ln_mlp_g, ln_mlp_b, sn2B, BS);
    gemm_kernel<<<dim3(D / 64, BS / 128), 256, GEMM_SMEM_BYTES, stream>>>(
        sn2B, W1B, mlp_b1, nullptr, nullptr, hB, D, 1);
    float* fOut = (it == 2) ? out : slotsF;
    gemm_kernel<<<dim3(D / 64, BS / 128), 256, GEMM_SMEM_BYTES, stream>>>(
        hB, W2B, mlp_b2, slotsF, fOut, slotsB, D, 0);
  }
}
